// VectorQuantizer_64544768524786
// MI455X (gfx1250) — matrix operand order repaired and checked by reading, not compiled, not run on hardware
//
#include <hip/hip_runtime.h>
#include <hip/hip_bf16.h>

#define VOCAB 1024
#define HIDDEN 64
#define NROWS (32 * 64 * 64)        // 131072 flattened vectors
#define HW 4096                     // 64*64 spatial
#define IMG_STRIDE (HIDDEN * HW)    // 262144 floats per image
#define QELEMS (32 * HIDDEN * HW)   // 8388608 quantized elements
#define WAVES_PER_BLOCK 8
#define ROWS_PER_WAVE 16

typedef __attribute__((ext_vector_type(16))) __bf16          v16bf;
typedef __attribute__((ext_vector_type(16))) unsigned short  v16us;
typedef __attribute__((ext_vector_type(8)))  unsigned short  v8us;
typedef __attribute__((ext_vector_type(8)))  float           v8f;
typedef __attribute__((ext_vector_type(4)))  float           v4f;
typedef __attribute__((ext_vector_type(2)))  float           v2f;

// f32 -> bf16 round-to-nearest-even (bit pattern; inputs are finite)
__device__ __forceinline__ unsigned short bf16_bits(float f) {
  unsigned u = __float_as_uint(f);
  u += 0x7FFFu + ((u >> 16) & 1u);
  return (unsigned short)(u >> 16);
}

// ---------------------------------------------------------------------------
// Kernel 1: per-call init. Codebook -> bf16 copy + f32 row norms; zero
// counts & sumsq (must re-zero every call: graph replay, no re-poison).
// ---------------------------------------------------------------------------
__global__ void vq_prep(const float* __restrict__ codebook,
                        unsigned int* __restrict__ counts,
                        float* __restrict__ sumsq,
                        float* __restrict__ cnorm,
                        unsigned short* __restrict__ cb16) {
  int v = blockIdx.x * blockDim.x + threadIdx.x;
  if (v == 0) *sumsq = 0.0f;
  if (v < VOCAB) {
    counts[v] = 0u;
    const float* row = codebook + (size_t)v * HIDDEN;
    unsigned short* dst = cb16 + (size_t)v * HIDDEN;
    float s = 0.0f;
#pragma unroll
    for (int k = 0; k < HIDDEN; ++k) {
      float f = row[k];
      s = fmaf(f, f, s);
      dst[k] = bf16_bits(f);
    }
    cnorm[v] = s;
  }
}

// ---------------------------------------------------------------------------
// Kernel 2: distances via bf16 WMMA, argmin, one_hot, quantized, loss partial.
// One wave owns a 16-row tile of flat X; 64 column tiles of 16 codes each.
// ---------------------------------------------------------------------------
__global__ void __launch_bounds__(256) vq_main(
    const float* __restrict__ inputs,          // [32][64][64][64] NCHW f32
    const float* __restrict__ codebook,        // [1024][64] f32
    const float* __restrict__ cnorm,           // [1024] f32
    const unsigned short* __restrict__ cb,     // [1024][64] bf16 bits
    float* __restrict__ q_out,                 // d_out + 1 (NCHW)
    float* __restrict__ one_hot,               // d_out + 2 + QELEMS
    unsigned int* __restrict__ counts,
    float* __restrict__ sumsq) {
  __shared__ float red_d[WAVES_PER_BLOCK][16][16];
  __shared__ int   red_i[WAVES_PER_BLOCK][16][16];
  __shared__ int   rowidx[WAVES_PER_BLOCK][16];
  __shared__ float blksum[256];

  const int tid  = threadIdx.x;
  const int w    = tid >> 5;
  const int lane = tid & 31;
  const int half = lane >> 4;   // which 16-lane half of the wave
  const int nl   = lane & 15;   // row (A) / column (B,C) residue

  const int rbase = (blockIdx.x * WAVES_PER_BLOCK + w) * ROWS_PER_WAVE;
  const int bimg  = rbase / HW;     // 16-row tile never crosses an image
  const int hwb   = rbase % HW;

  // ---- zero this wave's 16 one_hot rows (64 KB contiguous), nontemporal.
  // one_hot base is element-offset 8388610 == 2 (mod 4): region start is
  // 8B-aligned only. Peel an 8B head to reach 16B alignment, stream the body
  // as b128 stores (512B per wave-wide instruction), 8B tail.
  {
    float* reg = one_hot + (size_t)rbase * VOCAB;   // 16384 floats, byte%16==8
    if (lane == 0) {
      v2f z2 = {0.0f, 0.0f};
      __builtin_nontemporal_store(z2, (v2f*)reg);                      // [0,1]
      __builtin_nontemporal_store(z2, (v2f*)(reg + 16382));            // tail
    }
    v4f z4 = {0.0f, 0.0f, 0.0f, 0.0f};
    v4f* body = (v4f*)(reg + 2);                    // 16B aligned, 4095 quads
#pragma unroll 4
    for (int i = lane; i < 4095; i += 32)
      __builtin_nontemporal_store(z4, body + i);
  }

  // ---- A tile: rows rbase..rbase+15, K=0..63, NCHW gather -> bf16 in the
  // ISA 16-bit A layout: lane(m=nl,half): e<8 -> K=8h+e ; e>=8 -> K=16+8h+e-8
  const float* xb = inputs + (size_t)bimg * IMG_STRIDE + hwb + nl;
  v16bf a[2];
#pragma unroll
  for (int s = 0; s < 2; ++s) {
    v16us au;
#pragma unroll
    for (int j = 0; j < 8; ++j) {
      int k0 = 32 * s + 8 * half + j;
      int k1 = 32 * s + 16 + 8 * half + j;
      au[j]     = bf16_bits(xb[(size_t)k0 * HW]);   // coalesces across lanes 0-15
      au[8 + j] = bf16_bits(xb[(size_t)k1 * HW]);
    }
    a[s] = __builtin_bit_cast(v16bf, au);
  }

  // ---- per-lane running argmin over dist = ||c||^2 - 2 x.c  (||x||^2 cancels)
  float bd[8];
  int   bi[8];
#pragma unroll
  for (int j = 0; j < 8; ++j) { bd[j] = 3.4e38f; bi[j] = 0; }

  for (int ct = 0; ct < VOCAB / 16; ++ct) {
    const int n = ct * 16 + nl;                        // this lane's codebook row
    const v8us* brow = (const v8us*)(cb + (size_t)n * HIDDEN);
    v8f acc = {0.f, 0.f, 0.f, 0.f, 0.f, 0.f, 0.f, 0.f};
#pragma unroll
    for (int s = 0; s < 2; ++s) {
      // B layout: lane(n=nl,half): e<8 -> K = 8*half + e ; e>=8 -> K = 16 + 8*half + e-8 (+32s)
      v8us blo = brow[4 * s + half];
      v8us bhi = brow[4 * s + 2 + half];
      v16us bu;
#pragma unroll
      for (int j = 0; j < 8; ++j) { bu[j] = blo[j]; bu[8 + j] = bhi[j]; }
      v16bf bv = __builtin_bit_cast(v16bf, bu);
      acc = __builtin_amdgcn_wmma_f32_16x16x32_bf16(
          false, a[s], false, bv, (short)0, acc, false, false);
    }
    const float cn = cnorm[n];
#pragma unroll
    for (int j = 0; j < 8; ++j) {            // acc[j] = dot(row m=j+8*half, n)
      float d = fmaf(-2.0f, acc[j], cn);
      if (d < bd[j]) { bd[j] = d; bi[j] = n; }
    }
  }

  // ---- cross-lane argmin (16 candidates per row) via LDS
#pragma unroll
  for (int j = 0; j < 8; ++j) {
    int m = j + 8 * half;
    red_d[w][m][nl] = bd[j];
    red_i[w][m][nl] = bi[j];
  }
  __syncthreads();

  if (lane < 16) {
    float best = red_d[w][lane][0];
    int  besti = red_i[w][lane][0];
#pragma unroll
    for (int t = 1; t < 16; ++t) {
      float d = red_d[w][lane][t];
      int   i = red_i[w][lane][t];
      if (d < best || (d == best && i < besti)) { best = d; besti = i; }
    }
    rowidx[w][lane] = besti;
    const int r = rbase + lane;
    one_hot[(size_t)r * VOCAB + besti] = 1.0f;  // same-wave stores are in order
    atomicAdd(&counts[besti], 1u);
  }
  __syncthreads();

  // ---- quantized_st (== x + (q - x)) to NCHW + loss partial
  // (q_out base is only 4B aligned -> scalar stores; 16-lane 64B segments)
  float ls = 0.0f;
  const float* xq = inputs + (size_t)bimg * IMG_STRIDE + hwb;
  float* oq = q_out + (size_t)bimg * IMG_STRIDE + hwb;
#pragma unroll 4
  for (int i = lane; i < ROWS_PER_WAVE * HIDDEN; i += 32) {
    int m = i & 15;     // row in tile
    int c = i >> 4;     // channel
    float q = codebook[(size_t)rowidx[w][m] * HIDDEN + c];
    float x = xq[(size_t)c * HW + m];
    float st = x + (q - x);                 // mimic straight-through arithmetic
    __builtin_nontemporal_store(st, oq + (size_t)c * HW + m);
    float diff = q - x;
    ls = fmaf(diff, diff, ls);
  }
  blksum[tid] = ls;
  __syncthreads();
  for (int s = 128; s > 0; s >>= 1) {
    if (tid < s) blksum[tid] += blksum[tid + s];
    __syncthreads();
  }
  if (tid == 0) atomicAdd(sumsq, blksum[0]);
}

// ---------------------------------------------------------------------------
// Kernel 3: scalars. loss = (1+BETA)*mean((q-x)^2); perplexity from counts.
// ---------------------------------------------------------------------------
__global__ void vq_final(const unsigned int* __restrict__ counts,
                         const float* __restrict__ sumsq,
                         float* __restrict__ d_out) {
  __shared__ float sh[1024];
  int v = threadIdx.x;
  float p = (float)counts[v] * (1.0f / (float)NROWS);
  sh[v] = -p * logf(p + 1e-10f);
  __syncthreads();
  for (int s = 512; s > 0; s >>= 1) {
    if (v < s) sh[v] += sh[v + s];
    __syncthreads();
  }
  if (v == 0) {
    d_out[0] = sumsq[0] * (1.25f / (float)QELEMS);   // loss
    d_out[1 + QELEMS] = expf(sh[0]);                 // perplexity
  }
}

// ---------------------------------------------------------------------------
extern "C" void kernel_launch(void* const* d_in, const int* in_sizes, int n_in,
                              void* d_out, int out_size, void* d_ws, size_t ws_size,
                              hipStream_t stream) {
  (void)in_sizes; (void)n_in; (void)out_size; (void)ws_size;
  const float* inputs   = (const float*)d_in[0];
  const float* codebook = (const float*)d_in[1];

  float* out     = (float*)d_out;
  float* q_out   = out + 1;                    // [32,64,64,64]
  float* one_hot = out + 2 + QELEMS;           // [131072,1024]

  char* ws = (char*)d_ws;
  unsigned int*  counts = (unsigned int*)(ws);            //   4 KB
  float*         sumsq  = (float*)(ws + 4096);            //   4 B
  float*         cnormv = (float*)(ws + 8192);            //   4 KB
  unsigned short* cb16  = (unsigned short*)(ws + 12288);  // 128 KB

  vq_prep<<<4, 256, 0, stream>>>(codebook, counts, sumsq, cnormv, cb16);
  vq_main<<<NROWS / (WAVES_PER_BLOCK * ROWS_PER_WAVE), 256, 0, stream>>>(
      inputs, codebook, cnormv, cb16, q_out, one_hot, counts, sumsq);
  vq_final<<<1, 1024, 0, stream>>>(counts, sumsq, out);
}
